// TriPlaneDecoder_49254684950609
// MI455X (gfx1250) — compile-verified, matrix operand order checked by reading, not run on hardware
//
#include <hip/hip_runtime.h>

typedef __attribute__((ext_vector_type(16))) __bf16 v16bf;
typedef __attribute__((ext_vector_type(8)))  __bf16 v8bf;
typedef __attribute__((ext_vector_type(4)))  __bf16 v4bf;
typedef __attribute__((ext_vector_type(8)))  float  v8f;
typedef __attribute__((ext_vector_type(4)))  float  v4f;

#define B_      4
#define Q_      65536
#define LATENT  256
#define FEAT    32
#define RES     64
#define TRIDIM  (3 * FEAT * RES * RES)   /* 393216 */
#define NQ      (B_ * Q_)                /* 262144 */
#define TRITOT  (B_ * TRIDIM)           /* 1572864 */
#define LDST    272                      /* LDS row stride in bf16 elems (544B, 16B aligned) */
#define QT      128                      /* queries per workgroup */

// fast silu: bf16-precision pipeline -> v_rcp_f32 approximation is fine
__device__ __forceinline__ float silu_f(float v) {
  return v * __builtin_amdgcn_rcpf(1.0f + __expf(-v));
}

__device__ __forceinline__ v16bf mk16(v8bf lo, v8bf hi) {
  v16bf r;
#pragma unroll
  for (int i = 0; i < 8; i++) { r[i] = lo[i]; r[8 + i] = hi[i]; }
  return r;
}

// ---------------- Kernel 1: h = silu(z @ w1 + b1)  [4,512] ----------------
__global__ void k_latent(const float* __restrict__ z, const float* __restrict__ w1,
                         const float* __restrict__ b1, float* __restrict__ h) {
  int b = blockIdx.x, o = threadIdx.x;           // 4 blocks x 512 threads
  float acc = 0.f;
#pragma unroll 4
  for (int k = 0; k < LATENT; k++) acc += z[b * LATENT + k] * w1[k * (2 * LATENT) + o];
  acc += b1[o];
  h[b * (2 * LATENT) + o] = silu_f(acc);
}

// -------- Prep: f32 weights -> bf16 WMMA B-fragment layout (K padded) ------
// frag element id = ((kt*NT + nt)*32 + lane)*16 + e ; per-lane K pattern per ISA 7.12.2
__global__ void k_prep(const float* __restrict__ W, __bf16* __restrict__ dst,
                       int Kreal, int N) {
  int id = blockIdx.x * 256 + threadIdx.x;
  int e = id & 15, lane = (id >> 4) & 31, fi = id >> 9;
  int NT = N >> 4;
  int nt = fi % NT, kt = fi / NT;
  int k = kt * 32 + ((lane < 16) ? ((e < 8) ? e : e + 8) : ((e < 8) ? e + 8 : e + 16));
  int n = nt * 16 + (lane & 15);
  float v = (k < Kreal) ? W[k * N + n] : 0.f;
  dst[id] = (__bf16)v;
}

// -------- Kernel 2: triplanes = h @ w2 + b2 ; also transposed copy --------
// Memory-bound on w2 (805 MB): each thread owns 4 consecutive output cols,
// reads w2 exactly once, reuses each element across the 4 batches.
__global__ void k_triplane(const float* __restrict__ h, const float* __restrict__ w2,
                           const float* __restrict__ b2, float* __restrict__ out,
                           float* __restrict__ tp) {
  __shared__ float hl[B_ * 2 * LATENT];
  for (int i = threadIdx.x; i < B_ * 2 * LATENT; i += 256) hl[i] = h[i];
  __syncthreads();
  int j0 = (blockIdx.x * 256 + threadIdx.x) * 4;
  v4f acc[B_];
#pragma unroll
  for (int b = 0; b < B_; b++) acc[b] = (v4f){0.f, 0.f, 0.f, 0.f};
  for (int k = 0; k < 2 * LATENT; k++) {
    v4f w = *(const v4f*)(w2 + (size_t)k * TRIDIM + j0);
#pragma unroll
    for (int b = 0; b < B_; b++) acc[b] += hl[b * 512 + k] * w;
  }
  v4f bb = *(const v4f*)(b2 + j0);
  int p = j0 >> 17, rem = j0 & 131071, f = rem >> 12, yx = rem & 4095;
#pragma unroll
  for (int b = 0; b < B_; b++) {
    v4f r = acc[b] + bb;
    *(v4f*)(out + (size_t)b * TRIDIM + j0) = r;              // reference layout [B,3,F,R,R]
    float* tb = tp + ((size_t)(b * 3 + p) * 4096 + yx) * 32 + f;  // channel-last copy
#pragma unroll
    for (int c = 0; c < 4; c++) tb[(size_t)c * 32] = r[c];
  }
}

// ---------------- Fused sampler + MLP (WMMA bf16) ----------------
__device__ __forceinline__ v16bf load_a(const __bf16* in, int mtile, int kt, int lane) {
  int row = mtile * 16 + (lane & 15);
  int k0  = kt * 32 + ((lane < 16) ? 0 : 8);
  const __bf16* p = in + row * LDST + k0;
  return mk16(*(const v8bf*)p, *(const v8bf*)(p + 16));  // ds_load_b128 x2
}

template <int KT>
__device__ __forceinline__ void mlp_layer(const __bf16* in, __bf16* out,
                                          const __bf16* __restrict__ wf,
                                          const float* __restrict__ bias,
                                          int wave, int lane) {
  v16bf af[KT];
#pragma unroll
  for (int kt = 0; kt < KT; kt++) af[kt] = load_a(in, wave, kt, lane);
  int col  = lane & 15;
  int rofs = (lane < 16) ? 0 : 8;
#pragma unroll 1
  for (int nt = 0; nt < 16; nt++) {
    v8f acc = (v8f){0.f, 0.f, 0.f, 0.f, 0.f, 0.f, 0.f, 0.f};
#pragma unroll
    for (int kt = 0; kt < KT; kt++) {
      const __bf16* bp = wf + (size_t)((kt * 16 + nt) * 32 + lane) * 16;
      v16bf bfrag = mk16(*(const v8bf*)bp, *(const v8bf*)(bp + 8));  // global_load_b128 x2
      acc = __builtin_amdgcn_wmma_f32_16x16x32_bf16(false, af[kt], false, bfrag,
                                                    (short)0, acc, false, false);
    }
    float bi = bias[nt * 16 + col];
    __bf16* ob = out + (wave * 16 + rofs) * LDST + nt * 16 + col;
#pragma unroll
    for (int v = 0; v < 8; v++) {
      float x = acc[v] + bi;
      ob[v * LDST] = (__bf16)silu_f(x);
    }
  }
}

__global__ void __launch_bounds__(256) k_mlp(
    const float* __restrict__ qp, const float* __restrict__ tp,
    const __bf16* __restrict__ w0f, const __bf16* __restrict__ w1f,
    const __bf16* __restrict__ w2f,
    const float* __restrict__ mb0, const float* __restrict__ mb1,
    const float* __restrict__ mb2,
    const float* __restrict__ sdfw, const float* __restrict__ sdfb,
    const float* __restrict__ occw, const float* __restrict__ occb,
    float* __restrict__ out) {
  extern __shared__ __bf16 smem[];
  __bf16* bufA = smem;               // QT x LDST
  __bf16* bufB = smem + QT * LDST;   // QT x LDST
  int tid = threadIdx.x;
  int lane = tid & 31, wave = tid >> 5;
  int rbase = blockIdx.x * QT;

  // ---- bilinear sampling: 2 threads per query row (16 channels each) ----
  {
    int row = tid >> 1, half = tid & 1;
    int gr = rbase + row;
    int b = gr >> 16;
    float qx = qp[gr * 3 + 0], qy = qp[gr * 3 + 1], qz = qp[gr * 3 + 2];
    float cx = fminf(fmaxf(qx, -1.f), 1.f);
    float cy = fminf(fmaxf(qy, -1.f), 1.f);
    float cz = fminf(fmaxf(qz, -1.f), 1.f);
#pragma unroll
    for (int p = 0; p < 3; p++) {
      float sx = (p == 2) ? cy : cx;           // planes: (x,y) (x,z) (y,z)
      float sy = (p == 0) ? cy : cz;
      float fx = (sx + 1.f) * 0.5f * (RES - 1);
      float fy = (sy + 1.f) * 0.5f * (RES - 1);
      float x0f = floorf(fx), y0f = floorf(fy);
      float wx = fx - x0f, wy = fy - y0f;
      int x0 = (int)x0f; x0 = x0 < 0 ? 0 : (x0 > RES - 1 ? RES - 1 : x0);
      int y0 = (int)y0f; y0 = y0 < 0 ? 0 : (y0 > RES - 1 ? RES - 1 : y0);
      int x1 = x0 + 1 > RES - 1 ? RES - 1 : x0 + 1;
      int y1 = y0 + 1 > RES - 1 ? RES - 1 : y0 + 1;
      float w00 = (1.f - wx) * (1.f - wy), w01 = wx * (1.f - wy);
      float w10 = (1.f - wx) * wy,         w11 = wx * wy;
      const float* pb = tp + (size_t)(b * 3 + p) * 4096 * 32;
      int i00 = (y0 * RES + x0) * 32, i01 = (y0 * RES + x1) * 32;
      int i10 = (y1 * RES + x0) * 32, i11 = (y1 * RES + x1) * 32;
      int cbase = half * 16;
#pragma unroll
      for (int cc = 0; cc < 4; cc++) {
        int c = cbase + cc * 4;
        v4f a = *(const v4f*)(pb + i00 + c) * w00 + *(const v4f*)(pb + i01 + c) * w01 +
                *(const v4f*)(pb + i10 + c) * w10 + *(const v4f*)(pb + i11 + c) * w11;
        v4bf o;
#pragma unroll
        for (int e = 0; e < 4; e++) o[e] = (__bf16)a[e];
        *(v4bf*)(bufA + row * LDST + p * 32 + c) = o;
      }
    }
    if (half == 0) {  // raw (unclamped) coords + zero K-padding to 128
      __bf16* r = bufA + row * LDST;
      r[96] = (__bf16)qx; r[97] = (__bf16)qy; r[98] = (__bf16)qz;
#pragma unroll
      for (int k = 99; k < 128; k++) r[k] = (__bf16)0.f;
    }
  }
  __syncthreads();

  mlp_layer<4>(bufA, bufB, w0f, mb0, wave, lane);   // 99(->128) -> 256
  __syncthreads();
  mlp_layer<8>(bufB, bufA, w1f, mb1, wave, lane);   // 256 -> 256
  __syncthreads();
  mlp_layer<8>(bufA, bufB, w2f, mb2, wave, lane);   // 256 -> 256
  __syncthreads();

  // ---- heads: lanes 0-15 -> sdf, lanes 16-31 -> occ (one row per lane) ----
  // vectorized: ds_load_b128 for 8 bf16 activations + 2x global_load_b128 weights
  {
    int lrow = wave * 16 + (lane & 15);
    const __bf16* a = bufB + lrow * LDST;
    const float* w = (lane < 16) ? sdfw : occw;
    float acc = 0.f;
#pragma unroll 4
    for (int k = 0; k < 256; k += 8) {
      v8bf av = *(const v8bf*)(a + k);
      v4f wlo = *(const v4f*)(w + k);
      v4f whi = *(const v4f*)(w + k + 4);
#pragma unroll
      for (int e = 0; e < 4; e++) {
        acc += (float)av[e] * wlo[e];
        acc += (float)av[e + 4] * whi[e];
      }
    }
    acc += (lane < 16) ? sdfb[0] : occb[0];
    out[TRITOT + ((lane < 16) ? 0 : NQ) + rbase + lrow] = acc;
  }
}

// ------------------------------- launch -----------------------------------
extern "C" void kernel_launch(void* const* d_in, const int* in_sizes, int n_in,
                              void* d_out, int out_size, void* d_ws, size_t ws_size,
                              hipStream_t stream) {
  const float* z     = (const float*)d_in[0];
  const float* qp    = (const float*)d_in[1];
  const float* w1    = (const float*)d_in[2];
  const float* b1    = (const float*)d_in[3];
  const float* w2    = (const float*)d_in[4];
  const float* b2    = (const float*)d_in[5];
  const float* mw0   = (const float*)d_in[6];
  const float* mb0   = (const float*)d_in[7];
  const float* mw1   = (const float*)d_in[8];
  const float* mb1   = (const float*)d_in[9];
  const float* mw2   = (const float*)d_in[10];
  const float* mb2   = (const float*)d_in[11];
  const float* sdfw  = (const float*)d_in[12];
  const float* sdfb  = (const float*)d_in[13];
  const float* occw  = (const float*)d_in[14];
  const float* occb  = (const float*)d_in[15];
  float* out = (float*)d_out;

  char* ws = (char*)d_ws;
  const size_t H_OFF  = 0;                         // 4*512 f32        = 8 KB
  const size_t TP_OFF = 8192;                      // 4*3*4096*32 f32  = 6291456 B
  const size_t W0_OFF = TP_OFF + (size_t)6291456;  // 128*256 bf16     = 65536 B
  const size_t W1_OFF = W0_OFF + 65536;            // 256*256 bf16     = 131072 B
  const size_t W2_OFF = W1_OFF + 131072;
  float*  h   = (float*)(ws + H_OFF);
  float*  tp  = (float*)(ws + TP_OFF);
  __bf16* w0f = (__bf16*)(ws + W0_OFF);
  __bf16* w1f = (__bf16*)(ws + W1_OFF);
  __bf16* w2f = (__bf16*)(ws + W2_OFF);

  k_latent<<<B_, 2 * LATENT, 0, stream>>>(z, w1, b1, h);
  k_prep<<<(128 * 256) / 256, 256, 0, stream>>>(mw0, w0f, 99, 256);
  k_prep<<<(256 * 256) / 256, 256, 0, stream>>>(mw1, w1f, 256, 256);
  k_prep<<<(256 * 256) / 256, 256, 0, stream>>>(mw2, w2f, 256, 256);
  k_triplane<<<TRIDIM / (256 * 4), 256, 0, stream>>>(h, w2, b2, out, tp);

  size_t shbytes = (size_t)2 * QT * LDST * sizeof(__bf16);  // 139264 B
  k_mlp<<<NQ / QT, 256, shbytes, stream>>>(qp, tp, w0f, w1f, w2f, mb0, mb1, mb2,
                                           sdfw, sdfb, occw, occb, out);
}
